// MultiHeadAttention_16904991277459
// MI455X (gfx1250) — matrix operand order repaired and checked by reading, not compiled, not run on hardware
//
#include <hip/hip_runtime.h>
#include <hip/hip_bf16.h>

// ---------------------------------------------------------------------------
// Types for CDNA5 WMMA (wave32): v_wmma_f32_16x16x32_bf16
// ---------------------------------------------------------------------------
typedef __bf16 bf16_t;
typedef __attribute__((ext_vector_type(16))) __bf16 v16bf;
typedef __attribute__((ext_vector_type(8)))  __bf16 v8bf;
typedef __attribute__((ext_vector_type(8)))  float  v8f;

__device__ __forceinline__ bf16_t f2bf(float f) {
    union { float f; unsigned u; } x; x.f = f;
    unsigned r = x.u + 0x7FFFu + ((x.u >> 16) & 1u);   // round-to-nearest-even
    unsigned short h = (unsigned short)(r >> 16);
    union { unsigned short s; __bf16 b; } y; y.s = h;
    return y.b;
}

// A fragment (16x32, MxK): lane m=lane&15 holds A[m][kb..kb+15],
// kb=(lane>>4)*16.  `base` points at A[0][k0] (row-major).
__device__ __forceinline__ v16bf load_afrag(const bf16_t* base, int row_stride, int lane) {
    const bf16_t* p = base + (long)(lane & 15) * row_stride + ((lane >> 4) * 16);
    union { v16bf v; v8bf h[2]; } u;
    u.h[0] = *reinterpret_cast<const v8bf*>(p);
    u.h[1] = *reinterpret_cast<const v8bf*>(p + 8);
    return u.v;
}

// B fragment (32x16, KxN): lane holds column n=lane&15, K range
// (lane>>4)*16 .. +15.  `base` points at Bt[n0][k0] (N x K layout, K contiguous).
__device__ __forceinline__ v16bf load_bfrag(const bf16_t* base, int col_stride, int lane) {
    const bf16_t* p = base + (long)(lane & 15) * col_stride + ((lane >> 4) * 16);
    return *reinterpret_cast<const v16bf*>(p);
}

__device__ __forceinline__ v8f wmma_bf16(v16bf a, v16bf b, v8f c) {
    return __builtin_amdgcn_wmma_f32_16x16x32_bf16(false, a, false, b, (short)0, c, false, false);
}

// ---------------------------------------------------------------------------
// Problem constants
// ---------------------------------------------------------------------------
#define B_SZ   4
#define L_SZ   1024
#define H_NUM  16
#define D_HD   64
#define E_TOT  1024
#define M_TOT  (B_SZ * L_SZ)          // 4096

// Workspace layout (bytes)
#define OFF_X16   0L
#define OFF_WQT   (OFF_X16 + 4096L*1024*2)            //  8 MB
#define OFF_WKT   (OFF_WQT + 1024L*1024*2)
#define OFF_WVT   (OFF_WKT + 1024L*1024*2)
#define OFF_WOT   (OFF_WVT + 1024L*1024*2)
#define OFF_Q     (OFF_WOT + 64L*1024*2)
#define OFF_K     (OFF_Q   + 4096L*1024*2)
#define OFF_VT    (OFF_K   + 4096L*1024*2)
#define OFF_CTX   (OFF_VT  + 4096L*1024*2)

// ---------------------------------------------------------------------------
// 1) convert x (f32 -> bf16), row-major [4096,1024]
// ---------------------------------------------------------------------------
__global__ void cvt_x_kernel(const float* __restrict__ in, bf16_t* __restrict__ out) {
    int i = (blockIdx.x * 256 + threadIdx.x) * 4;
    #pragma unroll
    for (int j = 0; j < 4; ++j) out[i + j] = f2bf(in[i + j]);
}

// transpose + convert: in [K,N] f32 -> out [N,K] bf16
__global__ void tcvt_kernel(const float* __restrict__ in, bf16_t* __restrict__ out, int K, int N) {
    long tid = (long)blockIdx.x * 256 + threadIdx.x;   // over K*N
    int k = (int)(tid / N), n = (int)(tid % N);
    out[(long)n * K + k] = f2bf(in[(long)k * N + n]);
}

// ---------------------------------------------------------------------------
// 2) Projection GEMM: [4096,1024] @ [1024,1024] + bias, 64x64 block per wave
//    (4x4 register tiling: 1 b128 load per WMMA).
//    mode 0 -> Q [B,H,L,D], mode 1 -> K [B,H,L,D], mode 2 -> V^T [B,H,D,L]
// ---------------------------------------------------------------------------
__device__ __forceinline__ void store_proj_tile(const v8f& acc, int m0, int n0, int lane,
                                                const float* __restrict__ bias,
                                                bf16_t* __restrict__ out, int mode) {
    int n = n0 + (lane & 15);
    float bv = bias[n];
    int h = n >> 6, d = n & 63;
    #pragma unroll
    for (int i = 0; i < 8; ++i) {
        int m  = m0 + (lane >> 4) * 8 + i;   // global row = b*1024 + l
        int bb = m >> 10, l = m & 1023;
        float v = acc[i] + bv;
        long idx;
        if (mode == 2) idx = (((long)(bb * H_NUM + h)) * D_HD + d) * L_SZ + l;    // V^T
        else           idx = (((long)(bb * H_NUM + h)) * L_SZ + l) * D_HD + d;    // Q / K
        out[idx] = f2bf(v);
    }
}

__global__ void proj_kernel(const bf16_t* __restrict__ X, const bf16_t* __restrict__ WT,
                            const float* __restrict__ bias, bf16_t* __restrict__ out, int mode) {
    int wave = threadIdx.x >> 5, lane = threadIdx.x & 31;
    int t  = blockIdx.x * 8 + wave;        // 1024 blocks of 64x64 (64 x 16)
    int m0 = (t >> 4) * 64, n0 = (t & 15) * 64;
    const bf16_t* ap = X  + (long)m0 * 1024;
    const bf16_t* bp = WT + (long)n0 * 1024;
    v8f acc[4][4] = {};
    for (int k0 = 0; k0 < 1024; k0 += 32) {
        v16bf a[4], b[4];
        #pragma unroll
        for (int r = 0; r < 4; ++r) a[r] = load_afrag(ap + (long)(r * 16) * 1024 + k0, 1024, lane);
        #pragma unroll
        for (int c = 0; c < 4; ++c) b[c] = load_bfrag(bp + (long)(c * 16) * 1024 + k0, 1024, lane);
        #pragma unroll
        for (int r = 0; r < 4; ++r)
            #pragma unroll
            for (int c = 0; c < 4; ++c)
                acc[r][c] = wmma_bf16(a[r], b[c], acc[r][c]);
    }
    #pragma unroll
    for (int r = 0; r < 4; ++r)
        #pragma unroll
        for (int c = 0; c < 4; ++c)
            store_proj_tile(acc[r][c], m0 + r * 16, n0 + c * 16, lane, bias, out, mode);
}

// ---------------------------------------------------------------------------
// 3) Attention: one block (8 waves) per (batch, 32-query tile).
//    LDS: S f32 [32][1024] (128KB) + P bf16 [32][1024] (64KB) + reductions.
// ---------------------------------------------------------------------------
#define ATTN_LDS (32*1024*4 + 32*1024*2 + 32*8*4*2)

__global__ void attn_kernel(const bf16_t* __restrict__ Q, const bf16_t* __restrict__ K,
                            const bf16_t* __restrict__ Vt, const int* __restrict__ mask,
                            float* __restrict__ attn_out, bf16_t* __restrict__ ctx) {
    extern __shared__ char smem[];
    float*  S      = (float*)smem;                               // [32][1024]
    bf16_t* P      = (bf16_t*)(smem + 32*1024*4);                // [32][1024]
    float*  redmax = (float*)(smem + 32*1024*4 + 32*1024*2);     // [32][8]
    float*  redsum = redmax + 32*8;

    int tid = threadIdx.x, wave = tid >> 5, lane = tid & 31;
    int b  = blockIdx.x >> 5;          // 4 batches x 32 q-tiles = 128 blocks
    int q0 = (blockIdx.x & 31) * 32;
    const long hstride = (long)L_SZ * D_HD;

    for (int h = 0; h < H_NUM; ++h) {
        const bf16_t* Qh = Q  + (long)(b * H_NUM + h) * hstride;   // [1024][64]
        const bf16_t* Kh = K  + (long)(b * H_NUM + h) * hstride;   // [1024][64]
        const bf16_t* Vh = Vt + (long)(b * H_NUM + h) * hstride;   // [64][1024]

        // ---- phase 1: S = (Q Kt)/8 + mask.  Each wave holds all 4 Q frags
        //      (2 row-tiles x 2 d-halves); each loaded K fragment feeds 2 WMMAs.
        const bf16_t* qbase = Qh + (long)q0 * D_HD;
        v16bf aQ00 = load_afrag(qbase,                 D_HD, lane); // rows 0-15,  d 0-31
        v16bf aQ01 = load_afrag(qbase + 32,            D_HD, lane); // rows 0-15,  d 32-63
        v16bf aQ10 = load_afrag(qbase + 16L*D_HD,      D_HD, lane); // rows 16-31, d 0-31
        v16bf aQ11 = load_afrag(qbase + 16L*D_HD + 32, D_HD, lane); // rows 16-31, d 32-63
        for (int ct = wave; ct < 64; ct += 8) {        // 64 key tiles / 8 waves
            int key0 = ct * 16;
            const bf16_t* kbase = Kh + (long)key0 * D_HD;
            v16bf b0 = load_bfrag(kbase,      D_HD, lane);
            v16bf b1 = load_bfrag(kbase + 32, D_HD, lane);
            v8f c0 = {}, c1 = {};
            c0 = wmma_bf16(aQ00, b0, c0);
            c1 = wmma_bf16(aQ10, b0, c1);
            c0 = wmma_bf16(aQ01, b1, c0);
            c1 = wmma_bf16(aQ11, b1, c1);
            int col = key0 + (lane & 15);
            float mv = (mask[b * L_SZ + col] == 0) ? -__builtin_inff() : 0.0f;
            #pragma unroll
            for (int i = 0; i < 8; ++i) {
                int r = (lane >> 4) * 8 + i;
                S[r * 1024 + col]        = c0[i] * 0.125f + mv;
                S[(r + 16) * 1024 + col] = c1[i] * 0.125f + mv;
            }
        }
        __syncthreads();

        // ---- phase 2: exact softmax over each row of 1024 ----------------
        int row = tid >> 3, part = tid & 7;             // 32 rows x 8 parts
        float* srow = S + row * 1024 + part * 128;
        float mx = -__builtin_inff();
        for (int j = 0; j < 128; ++j) mx = fmaxf(mx, srow[j]);
        redmax[row * 8 + part] = mx;
        __syncthreads();
        mx = redmax[row * 8];
        #pragma unroll
        for (int j = 1; j < 8; ++j) mx = fmaxf(mx, redmax[row * 8 + j]);
        float sum = 0.f;
        for (int j = 0; j < 128; ++j) { float e = __expf(srow[j] - mx); srow[j] = e; sum += e; }
        redsum[row * 8 + part] = sum;
        __syncthreads();
        sum = 0.f;
        #pragma unroll
        for (int j = 0; j < 8; ++j) sum += redsum[row * 8 + j];
        float inv = 1.0f / sum;
        bf16_t* prow = P + row * 1024 + part * 128;
        float*  aout = attn_out + ((long)b * L_SZ + q0 + row) * L_SZ + part * 128;
        for (int j = 0; j < 128; ++j) {
            float p = srow[j] * inv;
            prow[j] = f2bf(p);
            float contrib = p * (1.0f / H_NUM);         // mean over heads
            if (h == 0) aout[j] = contrib; else aout[j] += contrib;
        }
        __syncthreads();

        // ---- phase 3: ctx_h = P @ V_h   [32x1024]@[1024x64] --------------
        int rt2 = wave >> 2, d0 = (wave & 3) * 16;      // 2x4 tiles, 1 per wave
        const bf16_t* pbase = P  + (long)(rt2 * 16) * 1024;
        const bf16_t* vbase = Vh + (long)d0 * 1024;
        v8f acc = {};
        for (int k0 = 0; k0 < 1024; k0 += 32) {
            v16bf a = load_afrag(pbase + k0, 1024, lane);   // from LDS
            v16bf v = load_bfrag(vbase + k0, 1024, lane);   // from V^T
            acc = wmma_bf16(a, v, acc);
        }
        int d = d0 + (lane & 15);
        #pragma unroll
        for (int i = 0; i < 8; ++i) {
            int r = q0 + rt2 * 16 + (lane >> 4) * 8 + i;
            ctx[((long)b * L_SZ + r) * E_TOT + h * D_HD + d] = f2bf(acc[i]);
        }
        __syncthreads();
    }
}

// ---------------------------------------------------------------------------
// 4) out = ctx @ Wo + bo : [4096,1024]@[1024,64] -> f32, 64x64 block per wave
// ---------------------------------------------------------------------------
__global__ void outproj_kernel(const bf16_t* __restrict__ CTX, const bf16_t* __restrict__ WoT,
                               const float* __restrict__ bo, float* __restrict__ out) {
    int wave = threadIdx.x >> 5, lane = threadIdx.x & 31;
    int t  = blockIdx.x * 8 + wave;       // 64 blocks of 64x64 (64 x 1)
    int m0 = t * 64;
    const bf16_t* ap = CTX + (long)m0 * 1024;
    const bf16_t* bp = WoT;               // all 64 output cols
    v8f acc[4][4] = {};
    for (int k0 = 0; k0 < 1024; k0 += 32) {
        v16bf a[4], b[4];
        #pragma unroll
        for (int r = 0; r < 4; ++r) a[r] = load_afrag(ap + (long)(r * 16) * 1024 + k0, 1024, lane);
        #pragma unroll
        for (int c = 0; c < 4; ++c) b[c] = load_bfrag(bp + (long)(c * 16) * 1024 + k0, 1024, lane);
        #pragma unroll
        for (int r = 0; r < 4; ++r)
            #pragma unroll
            for (int c = 0; c < 4; ++c)
                acc[r][c] = wmma_bf16(a[r], b[c], acc[r][c]);
    }
    #pragma unroll
    for (int r = 0; r < 4; ++r) {
        #pragma unroll
        for (int c = 0; c < 4; ++c) {
            int n = c * 16 + (lane & 15);
            float bv = bo[n];
            #pragma unroll
            for (int i = 0; i < 8; ++i) {
                int m = m0 + r * 16 + (lane >> 4) * 8 + i;
                out[(long)m * D_HD + n] = acc[r][c][i] + bv;
            }
        }
    }
}

// ---------------------------------------------------------------------------
extern "C" void kernel_launch(void* const* d_in, const int* in_sizes, int n_in,
                              void* d_out, int out_size, void* d_ws, size_t ws_size,
                              hipStream_t stream) {
    const float* x    = (const float*)d_in[0];
    const int*   mask = (const int*)d_in[1];
    const float* Wq   = (const float*)d_in[2];
    const float* bq   = (const float*)d_in[3];
    const float* Wk   = (const float*)d_in[4];
    const float* bk   = (const float*)d_in[5];
    const float* Wv   = (const float*)d_in[6];
    const float* bv   = (const float*)d_in[7];
    const float* Wo   = (const float*)d_in[8];
    const float* bo   = (const float*)d_in[9];

    char* ws = (char*)d_ws;
    bf16_t* X16 = (bf16_t*)(ws + OFF_X16);
    bf16_t* WQT = (bf16_t*)(ws + OFF_WQT);
    bf16_t* WKT = (bf16_t*)(ws + OFF_WKT);
    bf16_t* WVT = (bf16_t*)(ws + OFF_WVT);
    bf16_t* WOT = (bf16_t*)(ws + OFF_WOT);
    bf16_t* Qb  = (bf16_t*)(ws + OFF_Q);
    bf16_t* Kb  = (bf16_t*)(ws + OFF_K);
    bf16_t* VTb = (bf16_t*)(ws + OFF_VT);
    bf16_t* CTX = (bf16_t*)(ws + OFF_CTX);

    float* out_main = (float*)d_out;                 // [4,1024,64]
    float* out_attn = out_main + (long)M_TOT * D_HD; // [4,1024,1024]

    // 1) conversions / transposes
    cvt_x_kernel<<<(M_TOT * E_TOT) / (256 * 4), 256, 0, stream>>>(x, X16);
    tcvt_kernel<<<(1024 * 1024) / 256, 256, 0, stream>>>(Wq, WQT, 1024, 1024);
    tcvt_kernel<<<(1024 * 1024) / 256, 256, 0, stream>>>(Wk, WKT, 1024, 1024);
    tcvt_kernel<<<(1024 * 1024) / 256, 256, 0, stream>>>(Wv, WVT, 1024, 1024);
    tcvt_kernel<<<(1024 * 64) / 256,   256, 0, stream>>>(Wo, WOT, 1024, 64);

    // 2) Q / K / V projections (V stored transposed)
    proj_kernel<<<128, 256, 0, stream>>>(X16, WQT, bq, Qb, 0);
    proj_kernel<<<128, 256, 0, stream>>>(X16, WKT, bk, Kb, 1);
    proj_kernel<<<128, 256, 0, stream>>>(X16, WVT, bv, VTb, 2);

    // 3) attention (dynamic LDS ~194KB; CDNA5 WGP has 320KB)
    static_assert(ATTN_LDS <= 320 * 1024, "LDS budget");
    (void)hipFuncSetAttribute((const void*)attn_kernel,
                              hipFuncAttributeMaxDynamicSharedMemorySize, ATTN_LDS);
    attn_kernel<<<B_SZ * (L_SZ / 32), 256, ATTN_LDS, stream>>>(Qb, Kb, VTb, mask,
                                                               out_attn, CTX);

    // 4) output projection
    outproj_kernel<<<8, 256, 0, stream>>>(CTX, WOT, bo, out_main);
}
